// TokenGATClassifierSimple_88776974008769
// MI455X (gfx1250) — matrix operand order repaired and checked by reading, not compiled, not run on hardware
//
#include <hip/hip_runtime.h>
#include <hip/hip_bf16.h>
#include <math.h>

#define NNODES 40000
#define NEDGES 640000
#define EFULL  (NEDGES + NNODES)   // with self loops = 680000
#define DIM    128
#define NHEAD  4
#define CH     32
#define NGRAPH 512
#define NLAYER 3
#define NEG_SLOPE 0.2f
#define BN_EPS 1e-5f
#define WPAD   136                 // padded K-stride (halfs): 272B = 68 dwords -> conflict-free

typedef __attribute__((ext_vector_type(16))) _Float16 v16h;
typedef __attribute__((ext_vector_type(2)))  _Float16 v2h;
typedef __attribute__((ext_vector_type(8)))  float    v8f;

// ---------------------------------------------------------------------------
// float atomic max via signed/unsigned int atomics (valid with -inf init)
// ---------------------------------------------------------------------------
__device__ inline void atomicMaxFloat(float* addr, float v) {
    if (v >= 0.0f) atomicMax((int*)addr, __float_as_int(v));
    else           atomicMin((unsigned int*)addr, (unsigned int)__float_as_int(v));
}

// ---------------------------------------------------------------------------
// GEMM: xl = X @ Wl + bl ; xr = X @ Wr + br      (X: N x 128, W: 128 x 128)
// One wave per 16-node tile; 16 accumulator tiles (8 col-tiles x 2 matrices).
// B fragments double-buffered from transposed f16 LDS weights; A raw loads
// for the next k-block prefetched under the 16-WMMA tile loop.
// ---------------------------------------------------------------------------
__global__ __launch_bounds__(128) void gemm_xlxr_kernel(
    const float* __restrict__ X,
    const float* __restrict__ Wl, const float* __restrict__ bl,
    const float* __restrict__ Wr, const float* __restrict__ br,
    float* __restrict__ xl, float* __restrict__ xr)
{
    // W^T[n][k] padded: 2 * 128 * 136 halfs = 68 KB (320 KB/WGP available)
    __shared__ __align__(16) _Float16 sWT[2][DIM * WPAD];

    // transpose + f32->f16 convert, packing K-pairs into 4B LDS stores
    for (int t = threadIdx.x; t < (DIM * DIM) / 2; t += 128) {
        const int k2 = (t >> 7) << 1;     // even k
        const int n  = t & 127;
        v2h pl; pl[0] = (_Float16)Wl[k2 * DIM + n]; pl[1] = (_Float16)Wl[(k2 + 1) * DIM + n];
        *(v2h*)&sWT[0][n * WPAD + k2] = pl;
        v2h pr; pr[0] = (_Float16)Wr[k2 * DIM + n]; pr[1] = (_Float16)Wr[(k2 + 1) * DIM + n];
        *(v2h*)&sWT[1][n * WPAD + k2] = pr;
    }
    __syncthreads();

    const int wave    = threadIdx.x >> 5;
    const int lane    = threadIdx.x & 31;
    const int halfSel = lane >> 4;       // 0: lanes 0-15, 1: lanes 16-31
    const int ln      = lane & 15;
    const int tile    = blockIdx.x * 4 + wave;   // 2500 tiles exactly
    const int row0    = tile * 16;

    v8f acc[16];
    const v8f vzero = {0.f, 0.f, 0.f, 0.f, 0.f, 0.f, 0.f, 0.f};
#pragma unroll
    for (int t = 0; t < 16; ++t) acc[t] = vzero;

    const float* xrow = X + (size_t)(row0 + ln) * DIM;
    const _Float16* wlane = &sWT[0][ln * WPAD];   // + tile/mat offsets below

    // ---- A prefetch for kb = 0 ----
    const int kA0 = halfSel * 16;
    float4 r0 = *(const float4*)(xrow + kA0);
    float4 r1 = *(const float4*)(xrow + kA0 + 4);
    float4 r2 = *(const float4*)(xrow + kA0 + 8);
    float4 r3 = *(const float4*)(xrow + kA0 + 12);

#pragma unroll
    for (int kb = 0; kb < 4; ++kb) {
        // ---- convert prefetched A raw data to f16 fragment (contiguous k = 16h + s order) ----
        v16h a;
        a[0]  = (_Float16)r0.x; a[1]  = (_Float16)r0.y; a[2]  = (_Float16)r0.z; a[3]  = (_Float16)r0.w;
        a[4]  = (_Float16)r1.x; a[5]  = (_Float16)r1.y; a[6]  = (_Float16)r1.z; a[7]  = (_Float16)r1.w;
        a[8]  = (_Float16)r2.x; a[9]  = (_Float16)r2.y; a[10] = (_Float16)r2.z; a[11] = (_Float16)r2.w;
        a[12] = (_Float16)r3.x; a[13] = (_Float16)r3.y; a[14] = (_Float16)r3.z; a[15] = (_Float16)r3.w;

        // ---- issue A prefetch for kb+1 (hidden under the 16 WMMAs below) ----
        if (kb < 3) {
            const int kA = (kb + 1) * 32 + halfSel * 16;
            r0 = *(const float4*)(xrow + kA);
            r1 = *(const float4*)(xrow + kA + 4);
            r2 = *(const float4*)(xrow + kA + 8);
            r3 = *(const float4*)(xrow + kA + 12);
        }

        // ---- B tile loop, double-buffered LDS fragments ----
        const int kB = kb * 32 + halfSel * 16;
        v16h b0 = *(const v16h*)(wlane + kB);     // tile 0 (mat 0, n0 = 0)
        v16h b1;
#pragma unroll
        for (int t = 0; t < 16; ++t) {
            if (t < 15) {
                const int t1  = t + 1;
                const int off = ((t1 >> 3) * DIM + (t1 & 7) * 16) * WPAD + kB;
                b1 = *(const v16h*)(wlane + off);
            }
            acc[t] = __builtin_amdgcn_wmma_f32_16x16x32_f16(
                false, a, false, b0, (short)0, acc[t], false, false);
            if (t < 15) b0 = b1;
        }
    }

    // ---- epilogue: add bias, store ----
#pragma unroll
    for (int t = 0; t < 16; ++t) {
        const int mat = t >> 3;
        const int n   = (t & 7) * 16 + ln;
        const float bias = mat ? br[n] : bl[n];
        float* dst = (mat ? xr : xl) + n;
#pragma unroll
        for (int r = 0; r < 8; ++r) {
            const int row = row0 + r + halfSel * 8;
            dst[(size_t)row * DIM] = acc[t][r] + bias;
        }
    }
}

// ---------------------------------------------------------------------------
// Per-layer segment-state init: out_num = 0, m = -inf, denom = 0
// launched with exactly N*128 threads
// ---------------------------------------------------------------------------
__global__ void init_seg_kernel(float* __restrict__ onum,
                                float* __restrict__ m,
                                float* __restrict__ den)
{
    const int idx = blockIdx.x * 256 + threadIdx.x;
    onum[idx] = 0.0f;
    if (idx < NNODES * NHEAD) {
        m[idx]   = __int_as_float(0xFF800000);   // -inf
        den[idx] = 0.0f;
    }
}

__global__ void init_pool_kernel(float* __restrict__ pooled,
                                 float* __restrict__ cnt)
{
    const int idx = blockIdx.x * 256 + threadIdx.x;
    pooled[idx] = 0.0f;
    if (idx < NGRAPH) cnt[idx] = 0.0f;
}

// ---------------------------------------------------------------------------
// Edge pass A: logits[e,h] = att[h] . leaky_relu(xl[src]+xr[dst]); seg-max
// one thread per (edge, head);  xl/xr are L2-resident (20.5 MB each)
// ---------------------------------------------------------------------------
__global__ void edgeA_kernel(const float* __restrict__ xl,
                             const float* __restrict__ xr,
                             const int*   __restrict__ ei,
                             const float* __restrict__ att,
                             float* __restrict__ logits,
                             float* __restrict__ m)
{
    const int idx = blockIdx.x * 256 + threadIdx.x;
    if (idx >= EFULL * NHEAD) return;
    const int e  = idx >> 2;
    const int hh = idx & 3;
    int s, d;
    if (e < NEDGES) { s = ei[e]; d = ei[NEDGES + e]; }
    else            { s = e - NEDGES; d = s; }

    const float* pl = xl + (size_t)s * DIM + hh * CH;
    const float* pr = xr + (size_t)d * DIM + hh * CH;
    const float* pa = att + hh * CH;
    float acc = 0.0f;
#pragma unroll
    for (int c = 0; c < CH; ++c) {
        float v = pl[c] + pr[c];
        v = (v > 0.0f) ? v : NEG_SLOPE * v;
        acc += pa[c] * v;
    }
    logits[idx] = acc;
    atomicMaxFloat(&m[d * NHEAD + hh], acc);
}

// ---------------------------------------------------------------------------
// Edge pass B: a = exp(logit - m[dst]); denom[dst]+=a; out_num[dst]+=a*xl[src]
// normalization deferred to finalize (out = out_num/denom) -> one less pass
// ---------------------------------------------------------------------------
__global__ void edgeB_kernel(const float* __restrict__ xl,
                             const int*   __restrict__ ei,
                             const float* __restrict__ logits,
                             const float* __restrict__ m,
                             float* __restrict__ den,
                             float* __restrict__ onum)
{
    const int idx = blockIdx.x * 256 + threadIdx.x;
    if (idx >= EFULL * NHEAD) return;
    const int e  = idx >> 2;
    const int hh = idx & 3;
    int s, d;
    if (e < NEDGES) { s = ei[e]; d = ei[NEDGES + e]; }
    else            { s = e - NEDGES; d = s; }

    const float a = __expf(logits[idx] - m[d * NHEAD + hh]);
    atomicAdd(&den[d * NHEAD + hh], a);
    const float* pl = xl + (size_t)s * DIM + hh * CH;
    float*       po = onum + (size_t)d * DIM + hh * CH;
#pragma unroll
    for (int c = 0; c < CH; ++c) atomicAdd(&po[c], a * pl[c]);
}

// ---------------------------------------------------------------------------
// Finalize: h = relu(BN(out_num/denom + conv_bias)); last layer also pools
// one thread per (node, channel), exactly N*128 threads
// ---------------------------------------------------------------------------
__global__ void finalize_kernel(const float* __restrict__ onum,
                                const float* __restrict__ den,
                                const float* __restrict__ cb,
                                const float* __restrict__ gam,
                                const float* __restrict__ bet,
                                const float* __restrict__ mea,
                                const float* __restrict__ var,
                                float* __restrict__ hbuf,
                                int lastLayer,
                                const int* __restrict__ batch,
                                float* __restrict__ pooled,
                                float* __restrict__ cnt)
{
    const int idx = blockIdx.x * 256 + threadIdx.x;
    const int i  = idx >> 7;
    const int ch = idx & 127;
    const int hh = ch >> 5;
    float v = onum[idx] / den[i * NHEAD + hh] + cb[ch];
    v = (v - mea[ch]) * rsqrtf(var[ch] + BN_EPS) * gam[ch] + bet[ch];
    v = fmaxf(v, 0.0f);
    hbuf[idx] = v;
    if (lastLayer) {
        const int g = batch[i];
        atomicAdd(&pooled[g * DIM + ch], v);
        if (ch == 0) atomicAdd(&cnt[g], 1.0f);
    }
}

// ---------------------------------------------------------------------------
// Classifier: pooled mean -> 128->64 relu -> 64->2 ; one block per graph
// ---------------------------------------------------------------------------
__global__ __launch_bounds__(64) void mlp_kernel(
    const float* __restrict__ pooled, const float* __restrict__ cnt,
    const float* __restrict__ W1, const float* __restrict__ b1,
    const float* __restrict__ W2, const float* __restrict__ b2,
    float* __restrict__ out)
{
    __shared__ float sp[DIM];
    __shared__ float sh[64];
    const int g = blockIdx.x, t = threadIdx.x;
    const float invc = 1.0f / fmaxf(cnt[g], 1.0f);
    sp[t]      = pooled[g * DIM + t]      * invc;
    sp[t + 64] = pooled[g * DIM + 64 + t] * invc;
    __syncthreads();
    float acc = b1[t];
#pragma unroll 8
    for (int k = 0; k < DIM; ++k) acc += sp[k] * W1[k * 64 + t];
    sh[t] = fmaxf(acc, 0.0f);
    __syncthreads();
    if (t < 2) {
        float o = b2[t];
#pragma unroll 8
        for (int k = 0; k < 64; ++k) o += sh[k] * W2[k * 2 + t];
        out[g * 2 + t] = o;
    }
}

// ---------------------------------------------------------------------------
extern "C" void kernel_launch(void* const* d_in, const int* in_sizes, int n_in,
                              void* d_out, int out_size, void* d_ws, size_t ws_size,
                              hipStream_t stream)
{
    const float* x     = (const float*)d_in[0];
    const int*   ei    = (const int*)  d_in[1];
    const int*   batch = (const int*)  d_in[2];
    const float* Wl    = (const float*)d_in[3];
    const float* bl    = (const float*)d_in[4];
    const float* Wr    = (const float*)d_in[5];
    const float* br    = (const float*)d_in[6];
    const float* att   = (const float*)d_in[7];
    const float* cb    = (const float*)d_in[8];
    const float* gam   = (const float*)d_in[9];
    const float* bet   = (const float*)d_in[10];
    const float* mea   = (const float*)d_in[11];
    const float* var   = (const float*)d_in[12];
    const float* W1    = (const float*)d_in[13];
    const float* b1    = (const float*)d_in[14];
    const float* W2    = (const float*)d_in[15];
    const float* b2    = (const float*)d_in[16];
    float* out = (float*)d_out;

    float* ws     = (float*)d_ws;
    float* hbuf   = ws; ws += (size_t)NNODES * DIM;
    float* xl     = ws; ws += (size_t)NNODES * DIM;
    float* xr     = ws; ws += (size_t)NNODES * DIM;
    float* onum   = ws; ws += (size_t)NNODES * DIM;
    float* logits = ws; ws += (size_t)EFULL * NHEAD;
    float* m      = ws; ws += (size_t)NNODES * NHEAD;
    float* den    = ws; ws += (size_t)NNODES * NHEAD;
    float* pooled = ws; ws += (size_t)NGRAPH * DIM;
    float* cnt    = ws; ws += (size_t)NGRAPH;

    const int gemm_blocks = (NNODES / 16) / 4;               // 625
    const int node_blocks = (NNODES * DIM) / 256;            // 20000
    const int edge_blocks = (EFULL * NHEAD + 255) / 256;     // 10625
    const int pool_blocks = (NGRAPH * DIM) / 256;            // 256

    for (int i = 0; i < NLAYER; ++i) {
        const float* hin = (i == 0) ? x : hbuf;
        gemm_xlxr_kernel<<<gemm_blocks, 128, 0, stream>>>(
            hin, Wl + i * DIM * DIM, bl + i * DIM,
                 Wr + i * DIM * DIM, br + i * DIM, xl, xr);
        init_seg_kernel<<<node_blocks, 256, 0, stream>>>(onum, m, den);
        edgeA_kernel<<<edge_blocks, 256, 0, stream>>>(xl, xr, ei, att + i * DIM, logits, m);
        edgeB_kernel<<<edge_blocks, 256, 0, stream>>>(xl, ei, logits, m, den, onum);
        if (i == NLAYER - 1)
            init_pool_kernel<<<pool_blocks, 256, 0, stream>>>(pooled, cnt);
        finalize_kernel<<<node_blocks, 256, 0, stream>>>(
            onum, den, cb + i * DIM, gam + i * DIM, bet + i * DIM,
            mea + i * DIM, var + i * DIM, hbuf, (i == NLAYER - 1) ? 1 : 0,
            batch, pooled, cnt);
    }
    mlp_kernel<<<NGRAPH, 64, 0, stream>>>(pooled, cnt, W1, b1, W2, b2, out);
}
